// SE_Block_48490180772307
// MI455X (gfx1250) — matrix operand order repaired and checked by reading, not compiled, not run on hardware
//
#include <hip/hip_runtime.h>
#include <hip/hip_bf16.h>

typedef __attribute__((ext_vector_type(16))) _Float16 v16h;
typedef __attribute__((ext_vector_type(8)))  float    v8f;
typedef __attribute__((ext_vector_type(4)))  float    v4f;

// ---------------------------------------------------------------------------
// Kernel 1: global average pool per (b,h) row.  relation is [1024 rows x 65536]
// One 256-thread (8-wave) block per row; float4 coalesced loads; LDS tree
// reduction.  Reads the full 256 MB once -> bandwidth bound.
// ---------------------------------------------------------------------------
__global__ void se_gap_kernel(const float* __restrict__ rel,
                              float* __restrict__ gap) {
    __shared__ float sdata[256];
    const int row = blockIdx.x;  // 0..1023 == b*16 + h
    const v4f* p = (const v4f*)(rel + (size_t)row * 65536);
    float sum = 0.f;
    // 65536 floats = 16384 float4 per row; 64 iterations per thread
    for (int k = threadIdx.x; k < 16384; k += 256) {
        v4f v = p[k];
        sum += v.x + v.y + v.z + v.w;
    }
    sdata[threadIdx.x] = sum;
    __syncthreads();
    for (int s = 128; s > 0; s >>= 1) {
        if (threadIdx.x < s) sdata[threadIdx.x] += sdata[threadIdx.x + s];
        __syncthreads();
    }
    if (threadIdx.x == 0) gap[row] = sdata[0] * (1.0f / 65536.0f);
}

// ---------------------------------------------------------------------------
// Kernel 2: SE bottleneck MLP on the tensor core path.
// 1 block, 128 threads = 4 wave32s; wave w handles batches [16w, 16w+16).
// Layer 1: D1 = gap(16x16, K padded to 32) x W1^T(16x4, N padded to 16)
// Layer 2: D2 = relu(D1+b1)(16x4, K padded) x W2^T(4x16)
// scale = trunc-to-int(sigmoid(D2 + b2))  in {0,1}
// A-fragment layout (16-bit A 16x32): lane m=lane&15; elem e<8 -> K=(lane<16?0:8)+e
// B/C/D layout: N = lane&15;  D vgpr r -> M = r + (lane<16?0:8)
// ---------------------------------------------------------------------------
__global__ void se_mlp_kernel(const float* __restrict__ gap,
                              const float* __restrict__ W1,
                              const float* __restrict__ b1,
                              const float* __restrict__ W2,
                              const float* __restrict__ b2,
                              float* __restrict__ scale) {
    __shared__ float sbot[4][16][4];  // [wave][batch-in-tile][bottleneck c]

    const int wv     = threadIdx.x >> 5;   // 0..3
    const int lane   = threadIdx.x & 31;
    const int n      = lane & 15;          // N column for B/C/D fragments
    const int kbase  = (lane < 16) ? 0 : 8;
    const int mbase  = (lane < 16) ? 0 : 8;
    const int m16    = lane & 15;          // A-fragment row
    const int batch0 = wv * 16;

    // ---- layer 1 ----
    v16h a = {};                           // elems 8..15 stay 0 (K pad 16..31)
    #pragma unroll
    for (int e = 0; e < 8; ++e)
        a[e] = (_Float16)gap[(batch0 + m16) * 16 + (kbase + e)];

    v16h bm = {};                          // B[K=kbase+e][N=n] = W1[n][kbase+e], n<4 valid
    if (n < 4) {
        #pragma unroll
        for (int e = 0; e < 8; ++e)
            bm[e] = (_Float16)W1[n * 16 + (kbase + e)];
    }

    v8f c0 = {};
    v8f d1 = __builtin_amdgcn_wmma_f32_16x16x32_f16(
        false, a, false, bm, (short)0, c0, false, false);

    if (n < 4) {
        #pragma unroll
        for (int r = 0; r < 8; ++r) {
            float v = d1[r] + b1[n];
            sbot[wv][mbase + r][n] = v > 0.f ? v : 0.f;   // relu
        }
    }
    __syncthreads();

    // ---- layer 2 ----
    v16h a2 = {};                          // K = kbase+e, valid only K<4
    if (kbase == 0) {
        #pragma unroll
        for (int e = 0; e < 4; ++e)
            a2[e] = (_Float16)sbot[wv][m16][e];
    }
    v16h b2m = {};                         // B2[K=e][N=n] = W2[n][e]
    if (lane < 16) {
        #pragma unroll
        for (int e = 0; e < 4; ++e)
            b2m[e] = (_Float16)W2[n * 4 + e];
    }

    v8f c1 = {};
    v8f d2 = __builtin_amdgcn_wmma_f32_16x16x32_f16(
        false, a2, false, b2m, (short)0, c1, false, false);

    #pragma unroll
    for (int r = 0; r < 8; ++r) {
        float x = d2[r] + b2[n];
        float s = 1.0f / (1.0f + __expf(-x));             // sigmoid
        // faithful int64-truncation cast of the gate (0 unless s rounds to 1.0)
        scale[(batch0 + mbase + r) * 16 + n] = (float)(long long)s;
    }
}

// ---------------------------------------------------------------------------
// Kernel 3: out = relation * scale[row], streamed.  16M float4 elements.
// Non-temporal load + store: output is never re-read, input is last-use.
// row index = q >> 14 (16384 float4 per 65536-float row).
// ---------------------------------------------------------------------------
__global__ void se_apply_kernel(const float* __restrict__ rel,
                                const float* __restrict__ scale,
                                float* __restrict__ out) {
    const v4f* r4 = (const v4f*)rel;
    v4f* o4 = (v4f*)out;
    const int stride = gridDim.x * blockDim.x;
    for (int q = blockIdx.x * blockDim.x + threadIdx.x; q < (1 << 24); q += stride) {
        float s = scale[q >> 14];
        v4f v = __builtin_nontemporal_load(&r4[q]);
        v *= s;
        __builtin_nontemporal_store(v, &o4[q]);
    }
}

// ---------------------------------------------------------------------------
extern "C" void kernel_launch(void* const* d_in, const int* in_sizes, int n_in,
                              void* d_out, int out_size, void* d_ws, size_t ws_size,
                              hipStream_t stream) {
    const float* relation = (const float*)d_in[0];   // [64,16,256,256]
    const float* W1       = (const float*)d_in[1];   // [4,16]
    const float* b1       = (const float*)d_in[2];   // [4]
    const float* W2       = (const float*)d_in[3];   // [16,4]
    const float* b2       = (const float*)d_in[4];   // [16]
    float* out = (float*)d_out;

    float* gap   = (float*)d_ws;        // 1024 floats
    float* scale = gap + 1024;          // 1024 floats

    // 1) GAP: one block per (b,h) row
    se_gap_kernel<<<1024, 256, 0, stream>>>(relation, gap);
    // 2) SE MLP on WMMA: single block, 4 wave32s
    se_mlp_kernel<<<1, 128, 0, stream>>>(gap, W1, b1, W2, b2, scale);
    // 3) broadcast multiply, streaming
    se_apply_kernel<<<8192, 256, 0, stream>>>(relation, scale, out);
}
